// DDM_39857296507358
// MI455X (gfx1250) — compile-verified, matrix operand order checked by reading, not run on hardware
//
#include <hip/hip_runtime.h>
#include <hip/hip_bf16.h>

// ---------------- problem constants (from reference) ----------------
#define NN     50000
#define NPG    5000
#define EE     800000
#define HID    128
#define IN0    256          // IN_DIM + HID
#define NHEAD  4
#define SLOPE  0.2f

typedef __attribute__((ext_vector_type(16))) __bf16 v16bf;
typedef __attribute__((ext_vector_type(8)))  float  v8f;

// ---------------- workspace layout (bytes, all 256-aligned) ----------------
static const size_t OFF_H   = 0;                                // NN*256 f32 (later reused: o1 | out)
static const size_t OFF_HP  = OFF_H   + (size_t)NN*256*4;       // NN*256 bf16 packed-A fragments
static const size_t OFF_FS  = OFF_HP  + (size_t)NN*256*2;       // NN*128 f32
static const size_t OFF_FD  = OFF_FS  + (size_t)NN*128*4;       // NN*128 f32
static const size_t OFF_RST = OFF_FD  + (size_t)NN*128*4;       // NN*128 f32
static const size_t OFF_H1  = OFF_RST + (size_t)NN*128*4;       // NN*128 f32
static const size_t OFF_H2  = OFF_H1  + (size_t)NN*128*4;       // NN*128 f32
static const size_t OFF_EL  = OFF_H2  + (size_t)NN*128*4;       // EE*4  f32 (logits, then ee in place)
static const size_t OFF_MO  = OFF_EL  + (size_t)EE*4*4;         // NN*4  u32 (ordered segment max)
static const size_t OFF_DEN = OFF_MO  + (size_t)NN*4*4;         // NN*4  f32
static const size_t OFF_MU  = OFF_DEN + (size_t)NN*4*4;         // 128 f32
static const size_t OFF_SD  = OFF_MU  + 512;                    // 128 f32
static const size_t OFF_RED = OFF_SD  + 512;                    // 2 f32 loss accumulators
static const size_t OFF_WPK = OFF_RED + 256;                    // 131072 bf16 packed weights

// packed-weight element offsets inside OFF_WPK
#define WP_WS0 0
#define WP_WD0 32768
#define WP_WS1 65536
#define WP_WD1 81920
#define WP_WO1 98304
#define WP_WO2 114688

// ---------------- helpers ----------------
__device__ __forceinline__ float wsum(float v) {
#pragma unroll
  for (int off = 16; off > 0; off >>= 1) v += __shfl_xor(v, off, 32);
  return v;
}
__device__ __forceinline__ float lrelu(float v) { return (v > 0.f) ? v : SLOPE * v; }
__device__ __forceinline__ unsigned f2ord(float f) {
  unsigned u = __float_as_uint(f);
  return (u & 0x80000000u) ? ~u : (u | 0x80000000u);
}
__device__ __forceinline__ float ord2f(unsigned u) {
  u = (u & 0x80000000u) ? (u & 0x7FFFFFFFu) : ~u;
  return __uint_as_float(u);
}

// ============================================================================
// Fragment packing. WMMA 16-bit A-layout (ISA 7.12.2): lane l of a wave holds
// 16 bf16 elements; m = l&15, half = l>>4, element e -> k = ((e&8)<<1)+half*8+(e&7).
// We store each 16x32 K-step tile as 512 contiguous bf16 (lane-major, 32 B per
// lane) so the GEMM reads fragments with plain global_load_b128 pairs.
// ============================================================================

// A: [NN x K] f32 row-major -> packed fragments. One thread per (rowtile,kstep,lane).
template<int K>
__global__ __launch_bounds__(256)
void packA_kernel(const float* __restrict__ A, __bf16* __restrict__ P) {
  const int nks = K / 32;
  int idx = blockIdx.x * 256 + threadIdx.x;
  if (idx >= (NN / 16) * nks * 32) return;
  int lane = idx & 31;
  int kt   = (idx >> 5) % nks;
  int tm   = (idx >> 5) / nks;
  int m    = tm * 16 + (lane & 15);
  int half = lane >> 4;
  int kb   = kt * 32;
  v16bf t;
#pragma unroll
  for (int e = 0; e < 16; ++e) {
    int k = kb + ((e & 8) << 1) + half * 8 + (e & 7);   // two contiguous 8-float runs
    t[e] = (__bf16)A[(size_t)m * K + k];
  }
  *(v16bf*)(P + (size_t)idx * 16) = t;
}

// W: [K x 128] f32 row-major -> packed B fragments (n = lane&15, k spans as above).
template<int K>
__global__ __launch_bounds__(256)
void packW_kernel(const float* __restrict__ W, __bf16* __restrict__ P) {
  const int nks = K / 32;
  int idx = blockIdx.x * 256 + threadIdx.x;
  if (idx >= 8 * nks * 32) return;
  int lane = idx & 31;
  int kt   = (idx >> 5) % nks;
  int tn   = (idx >> 5) / nks;
  int n    = tn * 16 + (lane & 15);
  int half = lane >> 4;
  int kb   = kt * 32;
  v16bf t;
#pragma unroll
  for (int e = 0; e < 16; ++e) {
    int k = kb + ((e & 8) << 1) + half * 8 + (e & 7);
    t[e] = (__bf16)W[(size_t)k * HID + n];
  }
  *(v16bf*)(P + (size_t)idx * 16) = t;
}

// ============================================================================
// WMMA GEMM on packed fragments: C[NN x 128] = A[NN x K] @ W[K x 128].
// One wave computes a 16-row x 64-col strip: one A fragment feeds 4 B tiles
// per K-step -> 4x v_wmma_f32_16x16x32_bf16, fully unrolled over K.
// MODE: 0 = plain, 1 = +bias, 2 = +bias then ReLU.
// ============================================================================
template<int K, int MODE>
__global__ __launch_bounds__(256)
void gemm_wmma_kernel(const __bf16* __restrict__ Ap, const __bf16* __restrict__ Wp,
                      float* __restrict__ C, const float* __restrict__ bias) {
  const int nks = K / 32;
  int wave = threadIdx.x >> 5, lane = threadIdx.x & 31;
  int work = blockIdx.x * 8 + wave;              // (rowtile, colgroup) pairs: 3125*2
  if (work >= (NN / 16) * 2) return;
  int tm = work >> 1;
  int cg = work & 1;                              // which 64-col half
  const v16bf* Ab = (const v16bf*)Ap + (size_t)tm * nks * 32 + lane;
  const v16bf* Wb = (const v16bf*)Wp + lane;
  v8f a0 = {}, a1 = {}, a2 = {}, a3 = {};
#pragma unroll
  for (int kt = 0; kt < nks; ++kt) {
    v16bf av = Ab[(size_t)kt * 32];
    __builtin_prefetch(Ab + (size_t)(kt + 1) * 32, 0, 1);   // global_prefetch_b8
    v16bf b0 = Wb[(size_t)((cg * 4 + 0) * nks + kt) * 32];
    v16bf b1 = Wb[(size_t)((cg * 4 + 1) * nks + kt) * 32];
    v16bf b2 = Wb[(size_t)((cg * 4 + 2) * nks + kt) * 32];
    v16bf b3 = Wb[(size_t)((cg * 4 + 3) * nks + kt) * 32];
    a0 = __builtin_amdgcn_wmma_f32_16x16x32_bf16(false, av, false, b0, (short)0, a0, false, false);
    a1 = __builtin_amdgcn_wmma_f32_16x16x32_bf16(false, av, false, b1, (short)0, a1, false, false);
    a2 = __builtin_amdgcn_wmma_f32_16x16x32_bf16(false, av, false, b2, (short)0, a2, false, false);
    a3 = __builtin_amdgcn_wmma_f32_16x16x32_bf16(false, av, false, b3, (short)0, a3, false, false);
  }
  // D layout: VGPR r -> row half*8 + r, col = lane&15 within each 16-wide tile
  int half = lane >> 4;
  int n0 = cg * 64 + (lane & 15);
#pragma unroll
  for (int r = 0; r < 8; ++r) {
    int cm = tm * 16 + half * 8 + r;
    float* crow = C + (size_t)cm * HID + n0;
    float v0 = a0[r], v1 = a1[r], v2 = a2[r], v3 = a3[r];
    if (MODE >= 1) {
      v0 += bias[n0]; v1 += bias[n0 + 16]; v2 += bias[n0 + 32]; v3 += bias[n0 + 48];
    }
    if (MODE == 2) {
      v0 = fmaxf(v0, 0.f); v1 = fmaxf(v1, 0.f); v2 = fmaxf(v2, 0.f); v3 = fmaxf(v3, 0.f);
    }
    crow[0] = v0; crow[16] = v1; crow[32] = v2; crow[48] = v3;
  }
}

// ---------------- per-column mean / unbiased std of x ----------------
__global__ __launch_bounds__(256)
void colstats_kernel(const float* __restrict__ x, float* __restrict__ mu,
                     float* __restrict__ sd) {
  __shared__ float s1[256], s2[256];
  int j = blockIdx.x;
  float a = 0.f, b = 0.f;
  for (int r = threadIdx.x; r < NN; r += 256) {
    float v = x[(size_t)r * 128 + j];
    a += v; b += v * v;
  }
  s1[threadIdx.x] = a; s2[threadIdx.x] = b;
  __syncthreads();
  for (int st = 128; st > 0; st >>= 1) {
    if (threadIdx.x < st) {
      s1[threadIdx.x] += s1[threadIdx.x + st];
      s2[threadIdx.x] += s2[threadIdx.x + st];
    }
    __syncthreads();
  }
  if (threadIdx.x == 0) {
    float m = s1[0] / (float)NN;
    float var = (s2[0] - (float)NN * m * m) / (float)(NN - 1);  // ddof=1
    mu[j] = m;
    sd[j] = sqrtf(fmaxf(var, 0.f));
  }
}

// ---------------- sample_q + concat(x_t, temb) -> h[N x 256] ----------------
// one wave per node row; lane owns 4 contiguous dims (float4 loads/stores)
__global__ __launch_bounds__(256)
void build_h_kernel(const float* __restrict__ x, const float* __restrict__ noise,
                    const int* __restrict__ t, const float* __restrict__ sab,
                    const float* __restrict__ somab, const float* __restrict__ temb,
                    const float* __restrict__ mu, const float* __restrict__ sd,
                    float* __restrict__ h) {
  int row = blockIdx.x * 8 + (threadIdx.x >> 5);
  if (row >= NN) return;
  int lane = threadIdx.x & 31;
  int c0 = lane * 4;
  float4 nzv = *(const float4*)(noise + (size_t)row * 128 + c0);
  float nz[4] = {nzv.x, nzv.y, nzv.z, nzv.w};
  float s = nz[0] + nz[1] + nz[2] + nz[3];
  s = wsum(s) * (1.f / 128.f);
  float var = 0.f;
#pragma unroll
  for (int j = 0; j < 4; ++j) { float d = nz[j] - s; var += d * d; }
  var = wsum(var) * (1.f / 128.f);
  float rs = rsqrtf(var + 1e-5f);
  int tt = t[row / NPG];
  float s1 = sab[tt], s2 = somab[tt];
  float4 xv4  = *(const float4*)(x + (size_t)row * 128 + c0);
  float4 mu4  = *(const float4*)(mu + c0);
  float4 sd4  = *(const float4*)(sd + c0);
  float4 te4  = *(const float4*)(temb + (size_t)tt * 128 + c0);
  float xv[4] = {xv4.x, xv4.y, xv4.z, xv4.w};
  float mm[4] = {mu4.x, mu4.y, mu4.z, mu4.w};
  float ss[4] = {sd4.x, sd4.y, sd4.z, sd4.w};
  float o[4];
#pragma unroll
  for (int j = 0; j < 4; ++j) {
    float z  = (nz[j] - s) * rs * ss[j] + mm[j];           // LN(noise)*std + mu
    float sg = (xv[j] > 0.f) ? 1.f : ((xv[j] < 0.f) ? -1.f : 0.f);
    o[j] = s1 * xv[j] + s2 * sg * fabsf(z);                // sign(x)*|nz|
  }
  *(float4*)(h + (size_t)row * IN0 + c0)       = make_float4(o[0], o[1], o[2], o[3]);
  *(float4*)(h + (size_t)row * IN0 + 128 + c0) = te4;
}

// ---------------- GATv2 edge phase ----------------
// logits: one wave per edge; lane loads float4 (one global_load_b128 per row);
// lanes 8j..8j+7 cover head j; reduce within 8-lane groups.
__global__ __launch_bounds__(256)
void edge_logit_kernel(const float* __restrict__ fs, const float* __restrict__ fd,
                       const float* __restrict__ att, const int* __restrict__ src,
                       const int* __restrict__ dst, float* __restrict__ elog) {
  int e = blockIdx.x * 8 + (threadIdx.x >> 5);
  if (e >= EE) return;
  int lane = threadIdx.x & 31;
  int s = src[e], d = dst[e];
  float4 fsv = *(const float4*)(fs + (size_t)s * 128 + lane * 4);
  float4 fdv = *(const float4*)(fd + (size_t)d * 128 + lane * 4);
  float4 aa  = *(const float4*)(att + lane * 4);
  float p = lrelu(fsv.x + fdv.x) * aa.x + lrelu(fsv.y + fdv.y) * aa.y +
            lrelu(fsv.z + fdv.z) * aa.z + lrelu(fsv.w + fdv.w) * aa.w;
  p += __shfl_xor(p, 1, 32);
  p += __shfl_xor(p, 2, 32);
  p += __shfl_xor(p, 4, 32);
  if ((lane & 7) == 0) elog[(size_t)e * 4 + (lane >> 3)] = p;
}

__global__ __launch_bounds__(256)
void edge_max_kernel(const float* __restrict__ elog, const int* __restrict__ dst,
                     unsigned* __restrict__ mo) {
  int i = blockIdx.x * 256 + threadIdx.x;
  if (i >= EE * NHEAD) return;
  int e = i >> 2, hh = i & 3;
  int d = dst[e];
  atomicMax(&mo[(size_t)d * 4 + hh], f2ord(elog[i]));
}

__global__ __launch_bounds__(256)
void edge_exp_kernel(float* __restrict__ elog, const int* __restrict__ dst,
                     const unsigned* __restrict__ mo, float* __restrict__ den) {
  int i = blockIdx.x * 256 + threadIdx.x;
  if (i >= EE * NHEAD) return;
  int e = i >> 2, hh = i & 3;
  int d = dst[e];
  float m  = ord2f(mo[(size_t)d * 4 + hh]);
  float ee = expf(elog[i] - m);
  elog[i] = ee;                                   // reuse buffer as ee
  atomicAdd(&den[(size_t)d * 4 + hh], ee);
}

__global__ __launch_bounds__(256)
void edge_agg_kernel(const float* __restrict__ ee, const float* __restrict__ den,
                     const float* __restrict__ fs, const int* __restrict__ src,
                     const int* __restrict__ dst, float* __restrict__ rst) {
  int e = blockIdx.x * 8 + (threadIdx.x >> 5);
  if (e >= EE) return;
  int lane = threadIdx.x & 31;
  int s = src[e], d = dst[e];
  int hh = lane >> 3;                             // head owning dims 4*lane..4*lane+3
  float alpha = ee[(size_t)e * 4 + hh] / den[(size_t)d * 4 + hh];
  float4 fsv = *(const float4*)(fs + (size_t)s * 128 + lane * 4);
  float* dp = rst + (size_t)d * 128 + lane * 4;
  atomicAdd(dp + 0, alpha * fsv.x);
  atomicAdd(dp + 1, alpha * fsv.y);
  atomicAdd(dp + 2, alpha * fsv.z);
  atomicAdd(dp + 3, alpha * fsv.w);
}

// ---------------- gelu(+bias)(+residual)+LayerNorm, one wave per row ----------------
__global__ __launch_bounds__(256)
void post_kernel(const float* __restrict__ rst, const float* __restrict__ bias,
                 const float* __restrict__ resid, const float* __restrict__ g,
                 const float* __restrict__ be, float* __restrict__ out) {
  int row = blockIdx.x * 8 + (threadIdx.x >> 5);
  if (row >= NN) return;
  int lane = threadIdx.x & 31;
  int c0 = lane * 4;
  float4 rv = *(const float4*)(rst + (size_t)row * 128 + c0);
  float4 bv = *(const float4*)(bias + c0);
  float v[4] = {rv.x + bv.x, rv.y + bv.y, rv.z + bv.z, rv.w + bv.w};
  float s = 0.f;
#pragma unroll
  for (int j = 0; j < 4; ++j) {
    v[j] = 0.5f * v[j] * (1.f + erff(v[j] * 0.70710678118654752f));  // exact gelu
  }
  if (resid) {
    float4 hv = *(const float4*)(resid + (size_t)row * 128 + c0);
    v[0] += hv.x; v[1] += hv.y; v[2] += hv.z; v[3] += hv.w;
  }
  s = v[0] + v[1] + v[2] + v[3];
  s = wsum(s) * (1.f / 128.f);
  float var = 0.f;
#pragma unroll
  for (int j = 0; j < 4; ++j) { float d = v[j] - s; var += d * d; }
  var = wsum(var) * (1.f / 128.f);
  float rs = rsqrtf(var + 1e-5f);
  float4 gv = *(const float4*)(g + c0);
  float4 ev = *(const float4*)(be + c0);
  float4 o  = make_float4((v[0] - s) * rs * gv.x + ev.x, (v[1] - s) * rs * gv.y + ev.y,
                          (v[2] - s) * rs * gv.z + ev.z, (v[3] - s) * rs * gv.w + ev.w);
  *(float4*)(out + (size_t)row * 128 + c0) = o;
}

// ---------------- losses ----------------
__global__ __launch_bounds__(256)
void mse_kernel(const float* __restrict__ o, const float* __restrict__ x,
                float* __restrict__ acc) {
  __shared__ float sm[8];
  float a = 0.f;
  for (size_t i = (size_t)blockIdx.x * 256 + threadIdx.x; i < (size_t)NN * 32;
       i += (size_t)gridDim.x * 256) {
    float4 ov = ((const float4*)o)[i];
    float4 xv = ((const float4*)x)[i];
    float d0 = ov.x - xv.x, d1 = ov.y - xv.y, d2 = ov.z - xv.z, d3 = ov.w - xv.w;
    a += d0 * d0 + d1 * d1 + d2 * d2 + d3 * d3;
  }
  a = wsum(a);
  int lane = threadIdx.x & 31, wave = threadIdx.x >> 5;
  if (lane == 0) sm[wave] = a;
  __syncthreads();
  if (threadIdx.x == 0) {
    float t = 0.f;
    for (int w = 0; w < 8; ++w) t += sm[w];
    atomicAdd(acc, t);
  }
}

__global__ __launch_bounds__(256)
void cos_kernel(const float* __restrict__ o, const float* __restrict__ x,
                float* __restrict__ acc) {
  int row = blockIdx.x * 8 + (threadIdx.x >> 5);
  if (row >= NN) return;
  int lane = threadIdx.x & 31;
  float4 ov = *(const float4*)(o + (size_t)row * 128 + lane * 4);
  float4 xv = *(const float4*)(x + (size_t)row * 128 + lane * 4);
  float dt = ov.x * xv.x + ov.y * xv.y + ov.z * xv.z + ov.w * xv.w;
  float no = ov.x * ov.x + ov.y * ov.y + ov.z * ov.z + ov.w * ov.w;
  float nx = xv.x * xv.x + xv.y * xv.y + xv.z * xv.z + xv.w * xv.w;
  dt = wsum(dt); no = wsum(no); nx = wsum(nx);
  if (lane == 0) {
    float dn = fmaxf(sqrtf(no), 1e-12f) * fmaxf(sqrtf(nx), 1e-12f);
    float c = 1.f - dt / dn;
    atomicAdd(acc, c * c);                        // ALPHA_L = 2
  }
}

__global__ void fin_kernel(const float* __restrict__ red, float* __restrict__ out) {
  out[0] = red[0] / (float)((size_t)NN * 128) + 0.1f * (red[1] / (float)NN);
}

// ---------------- driver ----------------
extern "C" void kernel_launch(void* const* d_in, const int* in_sizes, int n_in,
                              void* d_out, int out_size, void* d_ws, size_t ws_size,
                              hipStream_t stream) {
  const float* x     = (const float*)d_in[0];
  const float* noise = (const float*)d_in[1];
  const int*   src   = (const int*)d_in[2];
  const int*   dst   = (const int*)d_in[3];
  const int*   t     = (const int*)d_in[4];
  const float* sab   = (const float*)d_in[5];
  const float* somab = (const float*)d_in[6];
  const float* temb  = (const float*)d_in[7];
  const float* Ws0   = (const float*)d_in[8];
  const float* Wd0   = (const float*)d_in[9];
  const float* a0    = (const float*)d_in[10];
  const float* b0    = (const float*)d_in[11];
  const float* g0    = (const float*)d_in[12];
  const float* be0   = (const float*)d_in[13];
  const float* Ws1   = (const float*)d_in[14];
  const float* Wd1   = (const float*)d_in[15];
  const float* a1    = (const float*)d_in[16];
  const float* b1    = (const float*)d_in[17];
  const float* g1    = (const float*)d_in[18];
  const float* be1   = (const float*)d_in[19];
  const float* Wo1   = (const float*)d_in[20];
  const float* bo1   = (const float*)d_in[21];
  const float* Wo2   = (const float*)d_in[22];
  const float* bo2   = (const float*)d_in[23];

  char*     ws  = (char*)d_ws;
  float*    h   = (float*)(ws + OFF_H);
  __bf16*   hp  = (__bf16*)(ws + OFF_HP);
  float*    fs  = (float*)(ws + OFF_FS);
  float*    fd  = (float*)(ws + OFF_FD);
  float*    rst = (float*)(ws + OFF_RST);
  float*    h1  = (float*)(ws + OFF_H1);
  float*    h2  = (float*)(ws + OFF_H2);
  float*    el  = (float*)(ws + OFF_EL);
  unsigned* mo  = (unsigned*)(ws + OFF_MO);
  float*    den = (float*)(ws + OFF_DEN);
  float*    mu  = (float*)(ws + OFF_MU);
  float*    sd  = (float*)(ws + OFF_SD);
  float*    red = (float*)(ws + OFF_RED);
  __bf16*   wp  = (__bf16*)(ws + OFF_WPK);
  float*    o1   = h;                        // reuse h region after layer-0 GEMMs
  float*    outb = h + (size_t)NN * 128;
  float*    dout = (float*)d_out;

  const int ROWB = (NN + 7) / 8;                       // 6250
  const int GEMB = ((NN / 16) * 2 + 7) / 8;            // 782 (16x64 strips)
  const int EDGB = (EE + 7) / 8;                       // 100000
  const int EHB  = (EE * NHEAD + 255) / 256;           // 12500
  const int PA256 = ((NN / 16) * 8 * 32 + 255) / 256;  // 3125
  const int PA128 = ((NN / 16) * 4 * 32 + 255) / 256;  // 1563

  // pack weights into WMMA fragment order (fp32 -> bf16)
  packW_kernel<IN0><<<8, 256, 0, stream>>>(Ws0, wp + WP_WS0);
  packW_kernel<IN0><<<8, 256, 0, stream>>>(Wd0, wp + WP_WD0);
  packW_kernel<HID><<<4, 256, 0, stream>>>(Ws1, wp + WP_WS1);
  packW_kernel<HID><<<4, 256, 0, stream>>>(Wd1, wp + WP_WD1);
  packW_kernel<HID><<<4, 256, 0, stream>>>(Wo1, wp + WP_WO1);
  packW_kernel<HID><<<4, 256, 0, stream>>>(Wo2, wp + WP_WO2);

  // sample_q + concat
  colstats_kernel<<<128, 256, 0, stream>>>(x, mu, sd);
  build_h_kernel<<<ROWB, 256, 0, stream>>>(x, noise, t, sab, somab, temb, mu, sd, h);

  // ---- GATv2 layer 0 ----
  packA_kernel<IN0><<<PA256, 256, 0, stream>>>(h, hp);
  gemm_wmma_kernel<IN0, 0><<<GEMB, 256, 0, stream>>>(hp, wp + WP_WS0, fs, nullptr);
  gemm_wmma_kernel<IN0, 0><<<GEMB, 256, 0, stream>>>(hp, wp + WP_WD0, fd, nullptr);
  hipMemsetAsync(mo,  0, (size_t)NN * 4 * 4, stream);
  hipMemsetAsync(den, 0, (size_t)NN * 4 * 4, stream);
  hipMemsetAsync(rst, 0, (size_t)NN * 128 * 4, stream);
  edge_logit_kernel<<<EDGB, 256, 0, stream>>>(fs, fd, a0, src, dst, el);
  edge_max_kernel<<<EHB, 256, 0, stream>>>(el, dst, mo);
  edge_exp_kernel<<<EHB, 256, 0, stream>>>(el, dst, mo, den);
  edge_agg_kernel<<<EDGB, 256, 0, stream>>>(el, den, fs, src, dst, rst);
  post_kernel<<<ROWB, 256, 0, stream>>>(rst, b0, nullptr, g0, be0, h1);

  // ---- GATv2 layer 1 (+residual) ----
  packA_kernel<HID><<<PA128, 256, 0, stream>>>(h1, hp);
  gemm_wmma_kernel<HID, 0><<<GEMB, 256, 0, stream>>>(hp, wp + WP_WS1, fs, nullptr);
  gemm_wmma_kernel<HID, 0><<<GEMB, 256, 0, stream>>>(hp, wp + WP_WD1, fd, nullptr);
  hipMemsetAsync(mo,  0, (size_t)NN * 4 * 4, stream);
  hipMemsetAsync(den, 0, (size_t)NN * 4 * 4, stream);
  hipMemsetAsync(rst, 0, (size_t)NN * 128 * 4, stream);
  edge_logit_kernel<<<EDGB, 256, 0, stream>>>(fs, fd, a1, src, dst, el);
  edge_max_kernel<<<EHB, 256, 0, stream>>>(el, dst, mo);
  edge_exp_kernel<<<EHB, 256, 0, stream>>>(el, dst, mo, den);
  edge_agg_kernel<<<EDGB, 256, 0, stream>>>(el, den, fs, src, dst, rst);
  post_kernel<<<ROWB, 256, 0, stream>>>(rst, b1, h1, g1, be1, h2);

  // ---- output MLP ----
  packA_kernel<HID><<<PA128, 256, 0, stream>>>(h2, hp);
  gemm_wmma_kernel<HID, 2><<<GEMB, 256, 0, stream>>>(hp, wp + WP_WO1, o1, bo1);   // relu(h@Wo1+bo1)
  packA_kernel<HID><<<PA128, 256, 0, stream>>>(o1, hp);
  gemm_wmma_kernel<HID, 1><<<GEMB, 256, 0, stream>>>(hp, wp + WP_WO2, outb, bo2); // @Wo2+bo2

  // ---- loss ----
  hipMemsetAsync(red, 0, 8, stream);
  mse_kernel<<<1024, 256, 0, stream>>>(outb, x, &red[0]);
  cos_kernel<<<ROWB, 256, 0, stream>>>(outb, x, &red[1]);
  fin_kernel<<<1, 1, 0, stream>>>(red, dout);
}
